// AttentionBasedRouter_51049981280909
// MI455X (gfx1250) — matrix operand order repaired and checked by reading, not compiled, not run on hardware
//
#include <hip/hip_runtime.h>
#include <hip/hip_bf16.h>

typedef __bf16 bf16_t;
typedef __attribute__((ext_vector_type(16))) __bf16 v16bf;
typedef __attribute__((ext_vector_type(8)))  float  v8f;
typedef __attribute__((ext_vector_type(4)))  unsigned int v4u;
typedef __attribute__((ext_vector_type(8)))  int v8i;
typedef __attribute__((ext_vector_type(4)))  int v4i;

constexpr int kB  = 2;
constexpr int kS  = 1024;
constexpr int kD  = 1024;
constexpr int kH  = 8;
constexpr int kDH = 128;
constexpr int kE  = 8;
constexpr int kQB = kS / 16;                     // 16-query blocks per (b,h)
constexpr float kScale = 0.08838834764831845f;   // 1/sqrt(dh)
constexpr float kInvHS = 1.0f / (float)(kH * kS);

struct alignas(16) BF8 { bf16_t h[8]; };

__device__ __forceinline__ v16bf zero_v16bf() {
  v16bf z;
#pragma unroll
  for (int i = 0; i < 16; ++i) z[i] = (bf16_t)0.0f;
  return z;
}

// A-matrix 16x32 bf16 tile, row-major source with leading dim `ld`.
__device__ __forceinline__ v16bf load_a16x32(const bf16_t* __restrict__ A, int ld) {
  const int lane  = threadIdx.x & 31;
  const int row   = lane & 15;
  const int khalf = (lane >> 4) << 4;            // 0 or 16
  const bf16_t* p = A + (size_t)row * ld + khalf;
  BF8 lo = *reinterpret_cast<const BF8*>(p);       // K = khalf .. khalf+7
  BF8 hi = *reinterpret_cast<const BF8*>(p + 8);   // K = khalf+8 .. khalf+15
  v16bf r;
#pragma unroll
  for (int i = 0; i < 8; ++i) { r[i] = lo.h[i]; r[i + 8] = hi.h[i]; }
  return r;
}

// B-matrix 32x16 (KxN) with B[k][n] = W[n][k] (NT): lane n reads a contiguous row chunk of W.
__device__ __forceinline__ v16bf load_bT32x16(const bf16_t* __restrict__ W, int ld,
                                              int n0, int k0) {
  const int lane = threadIdx.x & 31;
  const int n = n0 + (lane & 15);
  const int k = k0 + ((lane >> 4) << 4);          // 0 or 16
  const bf16_t* p = W + (size_t)n * ld + k;
  BF8 lo = *reinterpret_cast<const BF8*>(p);
  BF8 hi = *reinterpret_cast<const BF8*>(p + 8);
  v16bf r;
#pragma unroll
  for (int i = 0; i < 8; ++i) { r[i] = lo.h[i]; r[i + 8] = hi.h[i]; }
  return r;
}

__device__ __forceinline__ v8f wmma_bf16(v16bf a, v16bf b, v8f c) {
  return __builtin_amdgcn_wmma_f32_16x16x32_bf16(false, a, false, b, (short)0, c,
                                                 false, false);
}

// C/D layout: VGPR j, lanes 0-15 -> (M=j, N=lane); lanes 16-31 -> (M=j+8, N=lane-16).
__device__ __forceinline__ void store_c_bias_bf16(bf16_t* __restrict__ out, int ld, int n0,
                                                  v8f c, const float* __restrict__ bias) {
  const int lane  = threadIdx.x & 31;
  const int col   = n0 + (lane & 15);
  const int rbase = (lane >> 4) << 3;
  const float bv  = bias[col];
#pragma unroll
  for (int j = 0; j < 8; ++j)
    out[(size_t)(rbase + j) * ld + col] = (bf16_t)(c[j] + bv);
}

// ---------------- conversion kernels ----------------
__global__ __launch_bounds__(256) void k_cvt_bf16(const float* __restrict__ src,
                                                  bf16_t* __restrict__ dst, int n) {
  int i = blockIdx.x * 256 + threadIdx.x;
  if (i < n) dst[i] = (bf16_t)src[i];
}

// Wexp[e][d][f] (f32) -> WT[e][f][d] (bf16) so the routed GEMM is NT.
__global__ __launch_bounds__(256) void k_cvt_transpose(const float* __restrict__ src,
                                                       bf16_t* __restrict__ dst) {
  size_t i = (size_t)blockIdx.x * 256 + threadIdx.x;
  const size_t total = (size_t)kE * kD * kD;
  if (i >= total) return;
  int f = (int)(i % kD);
  size_t t = i / kD;
  int d = (int)(t % kD);
  int e = (int)(t / kD);
  dst[((size_t)e * kD + f) * kD + d] = (bf16_t)src[i];
}

// ================= TDM-staged q/k projection GEMM =================
// Workgroup = 4 waves. Tile: M=16 rows x N=64 cols. The W-operand (64 rows x 128 K)
// is staged to LDS by the Tensor Data Mover, double-buffered on TENSORcnt.
constexpr int kWRows   = 64;    // W rows staged per chunk
constexpr int kChunkK  = 128;   // K elements per chunk
constexpr int kLdsRowE = 136;   // bf16 elements per padded LDS row (256B data + 16B pad)

__device__ __forceinline__ void tdm_load_w_tile(const bf16_t* gsrc, unsigned lds_off) {
  // 2D tile: kWRows x kChunkK bf16 elements; tensor row stride kD elements.
  // LDS padding: after every 64 dwords (one 256B row) insert 4 dwords (16B).
  unsigned long long ga = (unsigned long long)(const void*)gsrc;
  v4u g0;
  g0[0] = 1u;                                        // count=1, user descriptor
  g0[1] = lds_off;                                   // LDS byte address
  g0[2] = (unsigned)(ga & 0xffffffffu);              // global_addr[31:0]
  g0[3] = (unsigned)(ga >> 32) | (2u << 30);         // global_addr[56:32] | type=2
  v8i g1;
  g1[0] = (int)((1u << 16) |                         // data_size = 2 bytes
                (1u << 20) |                         // pad_enable
                (5u << 22) |                         // pad_interval = 64 dwords
                (3u << 25));                         // pad_amount   = 4 dwords
  g1[1] = (int)(((unsigned)kD & 0xffffu) << 16);     // tensor_dim0[15:0] = K
  g1[2] = (int)((((unsigned)kD >> 16) & 0xffffu) |   // tensor_dim0[31:16]
                (((unsigned)kD & 0xffffu) << 16));   // tensor_dim1[15:0] = N rows
  g1[3] = (int)((((unsigned)kD >> 16) & 0xffffu) |   // tensor_dim1[31:16]
                ((unsigned)kChunkK << 16));          // tile_dim0 = 128
  g1[4] = kWRows;                                    // tile_dim1 = 64, tile_dim2 = 0
  g1[5] = kD;                                        // tensor_dim0_stride[31:0]
  g1[6] = 0;                                         // stride0 hi | stride1 lo
  g1[7] = 0;
  v4i z4 = {0, 0, 0, 0};
#if defined(__clang_major__) && (__clang_major__ >= 23)
  v8i z8 = {0, 0, 0, 0, 0, 0, 0, 0};
  __builtin_amdgcn_tensor_load_to_lds(g0, g1, z4, z4, z8, 0);
#else
  __builtin_amdgcn_tensor_load_to_lds(g0, g1, z4, z4, 0);
#endif
}

// B-tile read from the padded LDS staging buffer (ds_load_b128 pairs).
__device__ __forceinline__ v16bf load_bT_lds(const bf16_t* sW, int nbase, int k0) {
  const int lane = threadIdx.x & 31;
  const int n = nbase + (lane & 15);
  const int k = k0 + ((lane >> 4) << 4);
  const bf16_t* p = sW + (size_t)n * kLdsRowE + k;
  BF8 lo = *reinterpret_cast<const BF8*>(p);
  BF8 hi = *reinterpret_cast<const BF8*>(p + 8);
  v16bf r;
#pragma unroll
  for (int i = 0; i < 8; ++i) { r[i] = lo.h[i]; r[i + 8] = hi.h[i]; }
  return r;
}

__global__ __launch_bounds__(128) void k_gemm_nt_tdm(const bf16_t* __restrict__ A,
                                                     const bf16_t* __restrict__ W,
                                                     const float* __restrict__ bias,
                                                     bf16_t* __restrict__ C, int N, int K) {
  __shared__ bf16_t sW[2][kWRows * kLdsRowE];        // 2 x 64 x 272B = 34816 B
  const int n0   = blockIdx.x * 64;
  const int m0   = blockIdx.y * 16;
  const int wave = threadIdx.x >> 5;
  const bf16_t* Ab = A + (size_t)m0 * K;
  const int nChunks = K / kChunkK;

  if (wave == 0) {
    tdm_load_w_tile(W + (size_t)n0 * K, (unsigned)(size_t)(void*)&sW[0][0]);
    __builtin_amdgcn_s_wait_tensorcnt(0);
  }
  __syncthreads();

  v8f acc = {0.f, 0.f, 0.f, 0.f, 0.f, 0.f, 0.f, 0.f};
  for (int c = 0; c < nChunks; ++c) {
    if (wave == 0 && c + 1 < nChunks)
      tdm_load_w_tile(W + (size_t)n0 * K + (size_t)(c + 1) * kChunkK,
                      (unsigned)(size_t)(void*)&sW[(c + 1) & 1][0]);
    const bf16_t* sWc = &sW[c & 1][0];
    const int kg = c * kChunkK;
#pragma unroll
    for (int kk = 0; kk < 4; ++kk) {
      v16bf a  = load_a16x32(Ab + kg + kk * 32, K);
      v16bf bt = load_bT_lds(sWc, wave * 16, kk * 32);
      acc = wmma_bf16(a, bt, acc);
    }
    if (wave == 0 && c + 1 < nChunks)
      __builtin_amdgcn_s_wait_tensorcnt(0);        // next buffer landed
    __syncthreads();                               // all done reading current buffer
  }
  store_c_bias_bf16(C + (size_t)m0 * N, N, n0 + wave * 16, acc, bias);
}

// ---------------- per-expert q/k head projection: 16 rows x 128 cols, K=128 ----------------
__global__ __launch_bounds__(32) void k_expert_proj(const bf16_t* __restrict__ X,
                                                    const bf16_t* __restrict__ eW,
                                                    const float* __restrict__ eb,
                                                    bf16_t* __restrict__ Y) {
  const int s0 = blockIdx.x * 16;
  const int bh = blockIdx.y;
  const int b = bh >> 3, h = bh & 7;
  const int e = blockIdx.z;
  const bf16_t* Ab = X + ((size_t)(b * kS + s0)) * kD + h * kDH;
  const bf16_t* Wb = eW + (size_t)e * kDH * kDH;
  const float*  bb = eb + e * kDH;
  const size_t rowbase = (((size_t)e * kB + b) * kH + h) * kS;
  bf16_t* Yb = Y + (rowbase + s0) * kDH;

  v16bf a[4];
#pragma unroll
  for (int kk = 0; kk < 4; ++kk) a[kk] = load_a16x32(Ab + kk * 32, kD);

#pragma unroll
  for (int nt = 0; nt < 8; ++nt) {
    v8f acc = {0.f, 0.f, 0.f, 0.f, 0.f, 0.f, 0.f, 0.f};
#pragma unroll
    for (int kk = 0; kk < 4; ++kk) {
      v16bf bt = load_bT32x16(Wb, kDH, nt * 16, kk * 32);
      acc = wmma_bf16(a[kk], bt, acc);
    }
    store_c_bias_bf16(Yb, kDH, nt * 16, acc, bb);
  }
}

// ---------------- fused logits + softmax + per-query-block column sums ----------------
__global__ __launch_bounds__(256) void k_attn_scores(const bf16_t* __restrict__ QE,
                                                     const bf16_t* __restrict__ KE,
                                                     float* __restrict__ part) {
  __shared__ float lds[16][kS];   // 64 KB
  const int qb = blockIdx.x;
  const int bh = blockIdx.y;
  const int e  = blockIdx.z;
  const int b = bh >> 3, h = bh & 7;
  const int wave = threadIdx.x >> 5;
  const int lane = threadIdx.x & 31;

  const size_t rowbase = (((size_t)e * kB + b) * kH + h) * kS;
  const bf16_t* Qb = QE + (rowbase + (size_t)qb * 16) * kDH;
  const bf16_t* Kb = KE + rowbase * kDH;

  v16bf a[4];
#pragma unroll
  for (int kk = 0; kk < 4; ++kk) a[kk] = load_a16x32(Qb + kk * 32, kDH);

#pragma unroll
  for (int nt = 0; nt < 8; ++nt) {
    const int t0 = wave * 128 + nt * 16;
    v8f acc = {0.f, 0.f, 0.f, 0.f, 0.f, 0.f, 0.f, 0.f};
#pragma unroll
    for (int kk = 0; kk < 4; ++kk) {
      v16bf bt = load_bT32x16(Kb, kDH, t0, kk * 32);
      acc = wmma_bf16(a[kk], bt, acc);
    }
    const int col   = t0 + (lane & 15);
    const int rbase = (lane >> 4) << 3;
#pragma unroll
    for (int j = 0; j < 8; ++j) lds[rbase + j][col] = acc[j] * kScale;
  }
  __syncthreads();

  {
    const int row = threadIdx.x >> 4;
    const int seg = threadIdx.x & 15;
    float* rp = &lds[row][0];
    float m = -3.402823466e38f;
    for (int c = seg; c < kS; c += 16) m = fmaxf(m, rp[c]);
#pragma unroll
    for (int off = 8; off >= 1; off >>= 1) m = fmaxf(m, __shfl_xor(m, off, 16));
    float sum = 0.f;
    for (int c = seg; c < kS; c += 16) { float p = __expf(rp[c] - m); rp[c] = p; sum += p; }
#pragma unroll
    for (int off = 8; off >= 1; off >>= 1) sum += __shfl_xor(sum, off, 16);
    const float rinv = 1.0f / sum;
    for (int c = seg; c < kS; c += 16) rp[c] *= rinv;
  }
  __syncthreads();

  float* outp = part + ((((size_t)e * kB + b) * kH + h) * kQB + qb) * kS;
  for (int c = threadIdx.x; c < kS; c += 256) {
    float acc = 0.f;
#pragma unroll
    for (int r = 0; r < 16; ++r) acc += lds[r][c];
    outp[c] = acc * kInvHS;
  }
}

// ---------------- deterministic reduce: scores[token][e] = sum over (h, qb) ----------------
__global__ __launch_bounds__(256) void k_reduce_scores(const float* __restrict__ part,
                                                       float* __restrict__ scores) {
  const int idx = blockIdx.x * 256 + threadIdx.x;
  if (idx >= kB * kS * kE) return;
  const int e = idx & 7;
  const int bt = idx >> 3;
  const int b = bt >> 10, t = bt & 1023;
  float acc = 0.f;
  for (int h = 0; h < kH; ++h) {
    const float* ph = part + ((((size_t)e * kB + b) * kH + h) * kQB) * kS + t;
    for (int qb = 0; qb < kQB; ++qb) acc += ph[(size_t)qb * kS];
  }
  scores[idx] = acc;
}

// ---------------- routed expert transform: 16 x 64 tile per wave ----------------
__global__ __launch_bounds__(32) void k_routed(const bf16_t* __restrict__ X,
                                               const bf16_t* __restrict__ WT,
                                               const float* __restrict__ bexp,
                                               const float* __restrict__ scores,
                                               float* __restrict__ out) {
  const int n0 = blockIdx.x * 64;
  const int m0 = blockIdx.y * 16;
  const int lane = threadIdx.x;
  __shared__ int eids[16];
  if (lane < 16) {
    const float* srow = scores + (size_t)(m0 + lane) * kE;
    int best = 0; float bv = srow[0];
#pragma unroll
    for (int e = 1; e < kE; ++e) { float v = srow[e]; if (v > bv) { bv = v; best = e; } }
    eids[lane] = best;
  }
  __syncthreads();
  const int my_e = eids[lane & 15];
  const v16bf za = zero_v16bf();
  const bf16_t* Ab = X + (size_t)m0 * kD;

  v8f acc[4];
#pragma unroll
  for (int t = 0; t < 4; ++t) acc[t] = (v8f){0.f, 0.f, 0.f, 0.f, 0.f, 0.f, 0.f, 0.f};

  for (int e = 0; e < kE; ++e) {
    if (!__any(my_e == e)) continue;               // wave-uniform skip of absent experts
    const bool active = (my_e == e);
    const bf16_t* Wb = WT + (size_t)e * kD * kD;
    for (int k = 0; k < kD; k += 32) {
      if (k + 128 < kD) __builtin_prefetch(Wb + (size_t)n0 * kD + k + 128, 0, 1);
      v16bf a = za;
      if (active) a = load_a16x32(Ab + k, kD);     // one masked A load, reused 4x
#pragma unroll
      for (int t = 0; t < 4; ++t) {
        v16bf bt = load_bT32x16(Wb, kD, n0 + t * 16, k);
        acc[t] = wmma_bf16(a, bt, acc[t]);         // EXEC all-1s: wmma outside predicate
      }
    }
  }
  const int coll  = lane & 15;
  const int rbase = (lane >> 4) << 3;
#pragma unroll
  for (int t = 0; t < 4; ++t) {
    const int col = n0 + t * 16 + coll;
#pragma unroll
    for (int j = 0; j < 8; ++j) {
      const int r = rbase + j;
      out[(size_t)(m0 + r) * kD + col] = acc[t][j] + bexp[(size_t)eids[r] * kD + col];
    }
  }
}

// ---------------- host ----------------
extern "C" void kernel_launch(void* const* d_in, const int* in_sizes, int n_in,
                              void* d_out, int out_size, void* d_ws, size_t ws_size,
                              hipStream_t stream) {
  (void)in_sizes; (void)n_in; (void)out_size; (void)ws_size;
  const float* hs   = (const float*)d_in[0];
  const float* Wq   = (const float*)d_in[1];
  const float* bq   = (const float*)d_in[2];
  const float* Wk   = (const float*)d_in[3];
  const float* bk   = (const float*)d_in[4];
  const float* eWq  = (const float*)d_in[5];
  const float* ebq  = (const float*)d_in[6];
  const float* eWk  = (const float*)d_in[7];
  const float* ebk  = (const float*)d_in[8];
  const float* Wexp = (const float*)d_in[9];
  const float* bexp = (const float*)d_in[10];
  float* out = (float*)d_out;

  char* w = (char*)d_ws;
  auto take = [&](size_t bytes) -> void* {
    void* p = (void*)w; w += (bytes + 255) & ~(size_t)255; return p;
  };
  bf16_t* hs_b  = (bf16_t*)take((size_t)kB * kS * kD * 2);
  bf16_t* Wq_b  = (bf16_t*)take((size_t)kD * kD * 2);
  bf16_t* Wk_b  = (bf16_t*)take((size_t)kD * kD * 2);
  bf16_t* q_b   = (bf16_t*)take((size_t)kB * kS * kD * 2);
  bf16_t* k_b   = (bf16_t*)take((size_t)kB * kS * kD * 2);
  bf16_t* eWq_b = (bf16_t*)take((size_t)kE * kDH * kDH * 2);
  bf16_t* eWk_b = (bf16_t*)take((size_t)kE * kDH * kDH * 2);
  bf16_t* WT_b  = (bf16_t*)take((size_t)kE * kD * kD * 2);
  bf16_t* qe_b  = (bf16_t*)take((size_t)kE * kB * kH * kS * kDH * 2);
  bf16_t* ke_b  = (bf16_t*)take((size_t)kE * kB * kH * kS * kDH * 2);
  float*  part  = (float*)take((size_t)kE * kB * kH * kQB * kS * 4);
  float*  scr   = (float*)take((size_t)kB * kS * kE * 4);

  auto cvt = [&](const float* s, bf16_t* d, int n) {
    k_cvt_bf16<<<dim3((n + 255) / 256), dim3(256), 0, stream>>>(s, d, n);
  };
  cvt(hs,  hs_b,  kB * kS * kD);
  cvt(Wq,  Wq_b,  kD * kD);
  cvt(Wk,  Wk_b,  kD * kD);
  cvt(eWq, eWq_b, kE * kDH * kDH);
  cvt(eWk, eWk_b, kE * kDH * kDH);
  {
    size_t total = (size_t)kE * kD * kD;
    k_cvt_transpose<<<dim3((unsigned)((total + 255) / 256)), dim3(256), 0, stream>>>(Wexp, WT_b);
  }

  // q/k projections: TDM-staged NT GEMM (M = kB*kS, N = K = kD)
  k_gemm_nt_tdm<<<dim3(kD / 64, (kB * kS) / 16), dim3(128), 0, stream>>>(hs_b, Wq_b, bq, q_b, kD, kD);
  k_gemm_nt_tdm<<<dim3(kD / 64, (kB * kS) / 16), dim3(128), 0, stream>>>(hs_b, Wk_b, bk, k_b, kD, kD);

  // per-expert head projections
  k_expert_proj<<<dim3(kS / 16, kB * kH, kE), dim3(32), 0, stream>>>(q_b, eWq_b, ebq, qe_b);
  k_expert_proj<<<dim3(kS / 16, kB * kH, kE), dim3(32), 0, stream>>>(k_b, eWk_b, ebk, ke_b);

  // fused attention logits + softmax + per-block means
  k_attn_scores<<<dim3(kQB, kB * kH, kE), dim3(256), 0, stream>>>(qe_b, ke_b, part);

  // deterministic score reduction
  k_reduce_scores<<<dim3((kB * kS * kE) / 256), dim3(256), 0, stream>>>(part, scr);

  // top-1 routed expert transform (argmax folded in)
  k_routed<<<dim3(kD / 64, (kB * kS) / 16), dim3(32), 0, stream>>>(hs_b, WT_b, bexp, scr, out);
}
